// GCN_51694226375165
// MI455X (gfx1250) — matrix operand order repaired and checked by reading, not compiled, not run on hardware
//
#include <hip/hip_runtime.h>
#include <hip/hip_bf16.h>

// ---------------------------------------------------------------------------
// GCN (4 tiny conv layers, CSR-by-destination aggregation, L2-resident) +
// FC stack via bf16 WMMA (v_wmma_f32_16x16x32_bf16).
//   - A pre-packed to per-lane WMMA bf16 layout [kt][mt][lane], zero-padded
//   - one wave owns all 4 M-tiles: B tile loaded ONCE per k-step, 4 WMMAs
//     (minimum possible HBM traffic on the 320MB FC1 weight matrix)
//   - software-pipelined (double-buffered) guard-free main loop; guarded
//     code only for K/N tails
//   - split-K fp32 atomic reduction into bias-initialized output
// ---------------------------------------------------------------------------

#define N_NODES   20000
#define N_EDGES   640000
#define E_TOT     (N_EDGES + N_NODES)   // edges + self loops
#define BATCH     64
#define CH        4
#define NODE_STRIDE (BATCH * CH)        // 256 floats per node
#define NUM_CLASSES 150

typedef __attribute__((ext_vector_type(16))) __bf16 v16bf;
typedef __attribute__((ext_vector_type(8)))  float  v8f;

// ---------------------------------------------------------------------------
// CSR construction
// ---------------------------------------------------------------------------
__global__ void k_init(int* __restrict__ cnt, int* __restrict__ cursor) {
    int n = blockIdx.x * blockDim.x + threadIdx.x;
    if (n < N_NODES) { cnt[n] = 1; cursor[n] = 0; }   // 1 = self loop
}

__global__ void k_count(const long long* __restrict__ edge, int* __restrict__ cnt) {
    int e = blockIdx.x * blockDim.x + threadIdx.x;
    if (e < N_EDGES) {
        int col = (int)edge[N_EDGES + e];             // target
        atomicAdd(&cnt[col], 1);
    }
}

__global__ void k_dinv(const int* __restrict__ cnt, float* __restrict__ dinv) {
    int n = blockIdx.x * blockDim.x + threadIdx.x;
    if (n < N_NODES) dinv[n] = rsqrtf((float)cnt[n]); // deg >= 1 always
}

// single-block exclusive scan of 20000 counts -> offs[20001]
__global__ void k_scan(const int* __restrict__ cnt, int* __restrict__ offs) {
    __shared__ int s[1024];
    const int CHUNK = (N_NODES + 1023) / 1024;        // 20
    int t = threadIdx.x;
    int base = t * CHUNK;
    int sum = 0;
    for (int i = 0; i < CHUNK; ++i) {
        int idx = base + i;
        if (idx < N_NODES) sum += cnt[idx];
    }
    s[t] = sum;
    __syncthreads();
    for (int d = 1; d < 1024; d <<= 1) {
        int v = (t >= d) ? s[t - d] : 0;
        __syncthreads();
        s[t] += v;
        __syncthreads();
    }
    int run = s[t] - sum;                              // exclusive prefix
    for (int i = 0; i < CHUNK; ++i) {
        int idx = base + i;
        if (idx < N_NODES) { offs[idx] = run; run += cnt[idx]; }
    }
    if (t == 0) offs[N_NODES] = E_TOT;
}

__global__ void k_fill(const long long* __restrict__ edge,
                       const int* __restrict__ offs, int* __restrict__ cursor,
                       const float* __restrict__ dinv,
                       int* __restrict__ csr_src, float* __restrict__ csr_w) {
    int e = blockIdx.x * blockDim.x + threadIdx.x;
    if (e >= E_TOT) return;
    int row, col;
    if (e < N_EDGES) { row = (int)edge[e]; col = (int)edge[N_EDGES + e]; }
    else             { row = col = e - N_EDGES; }     // self loop
    int pos = offs[col] + atomicAdd(&cursor[col], 1);
    csr_src[pos] = row;
    csr_w[pos]   = dinv[row] * dinv[col];
}

// ---------------------------------------------------------------------------
// GCN conv: linear transforms (tiny) + gather aggregation (no float atomics)
// Node feature layout: h[node][batch][channel], 256 contiguous floats/node.
// ---------------------------------------------------------------------------
__global__ void k_lin1(const float* __restrict__ x, const float* __restrict__ W,
                       float* __restrict__ out) {
    int idx = blockIdx.x * blockDim.x + threadIdx.x;   // n*BATCH + b
    if (idx >= N_NODES * BATCH) return;
    int n = idx / BATCH, b = idx % BATCH;
    float v = x[(size_t)b * N_NODES + n];              // x is [B, N, 1]
    float4 o = { v * W[0], v * W[1], v * W[2], v * W[3] };
    *(float4*)(out + (size_t)idx * 4) = o;
}

__global__ void k_lin44(const float* __restrict__ in, const float* __restrict__ W,
                        float* __restrict__ out) {
    int idx = blockIdx.x * blockDim.x + threadIdx.x;
    if (idx >= N_NODES * BATCH) return;
    const float4 h = *(const float4*)(in + (size_t)idx * 4);
    float4 o;
    o.x = h.x*W[0] + h.y*W[4] + h.z*W[8]  + h.w*W[12];
    o.y = h.x*W[1] + h.y*W[5] + h.z*W[9]  + h.w*W[13];
    o.z = h.x*W[2] + h.y*W[6] + h.z*W[10] + h.w*W[14];
    o.w = h.x*W[3] + h.y*W[7] + h.z*W[11] + h.w*W[15];
    *(float4*)(out + (size_t)idx * 4) = o;
}

// one block per destination node; 256 threads = one (b,c) element each.
__global__ void k_agg(const float* __restrict__ hIn,
                      const int* __restrict__ csr_src, const float* __restrict__ csr_w,
                      const int* __restrict__ offs, const float* __restrict__ bias,
                      float* __restrict__ hOut) {
    __shared__ int   s_src[64];
    __shared__ float s_w[64];
    int n   = blockIdx.x;
    int tid = threadIdx.x;                   // b*4 + c
    int beg = offs[n], end = offs[n + 1];
    float acc = bias[tid & 3];
    for (int e0 = beg; e0 < end; e0 += 64) {
        int m = min(64, end - e0);
        if (tid < m) { s_src[tid] = csr_src[e0 + tid]; s_w[tid] = csr_w[e0 + tid]; }
        __syncthreads();
        for (int j = 0; j < m; ++j)
            acc += s_w[j] * hIn[(size_t)s_src[j] * NODE_STRIDE + tid];
        __syncthreads();
    }
    hOut[(size_t)n * NODE_STRIDE + tid] = fmaxf(acc, 0.0f);  // fused bias+relu
}

// ---------------------------------------------------------------------------
// A-matrix packing, layout [kt][mt][lane]: tile (kt,mt) = 1024B, lane l owns
// 16 bf16 (32B) at Ap + ((kt*4 + mt)*32 + l)*16.  Per ISA A layout:
// lane&15 = M row; lanes<16 hold K pairs {0..7,16..23}, lanes>=16 hold
// {8..15,24..31} (consecutive-K pairs per dword).
// ---------------------------------------------------------------------------
__device__ __forceinline__ int a_kidx(int kb, int p) {
    return (p < 4) ? (kb + 2 * p) : (16 + kb + 2 * (p - 4));
}

// FC1: fused reshape [n][b][c] -> A[b][n*4+c] + bf16 pack (K = 80000 exact)
__global__ void k_apack_gcn(const float* __restrict__ h, __bf16* __restrict__ Ap,
                            int nkt) {
    int tid = blockIdx.x * blockDim.x + threadIdx.x;
    if (tid >= nkt * 128) return;
    int lane = tid & 31;
    int mt   = (tid >> 5) & 3;
    int kt   = tid >> 7;
    int m    = mt * 16 + (lane & 15);                   // batch index
    int kb   = (lane < 16) ? 0 : 8;
    __bf16* out = Ap + ((size_t)tid << 4);
#pragma unroll
    for (int p = 0; p < 8; ++p) {
        int kk = (kt << 5) + a_kidx(kb, p);             // even; kk+1 same node
        float f0 = h[(size_t)(kk >> 2) * NODE_STRIDE + m * 4 + (kk & 3)];
        float f1 = h[(size_t)((kk + 1) >> 2) * NODE_STRIDE + m * 4 + ((kk + 1) & 3)];
        out[2 * p]     = (__bf16)f0;
        out[2 * p + 1] = (__bf16)f1;
    }
}

// generic: pack fp32 row-major A[64, K] (zero-pad K tail)
__global__ void k_apack(const float* __restrict__ A, __bf16* __restrict__ Ap,
                        int K, int nkt) {
    int tid = blockIdx.x * blockDim.x + threadIdx.x;
    if (tid >= nkt * 128) return;
    int lane = tid & 31;
    int mt   = (tid >> 5) & 3;
    int kt   = tid >> 7;
    int m    = mt * 16 + (lane & 15);
    int kb   = (lane < 16) ? 0 : 8;
    const float* Arow = A + (size_t)m * K;
    __bf16* out = Ap + ((size_t)tid << 4);
#pragma unroll
    for (int p = 0; p < 8; ++p) {
        int kk = (kt << 5) + a_kidx(kb, p);
        float f0 = (kk     < K) ? Arow[kk]     : 0.0f;
        float f1 = (kk + 1 < K) ? Arow[kk + 1] : 0.0f;
        out[2 * p]     = (__bf16)f0;
        out[2 * p + 1] = (__bf16)f1;
    }
}

// ---------------------------------------------------------------------------
// FC GEMM: Out(64,N) += A(64,K)*W(K,N).  One wave per (n-tile, k-chunk) owns
// all 4 M-tiles -> B loaded once per k-step, 4 WMMAs.  Double-buffered.
// ---------------------------------------------------------------------------
__global__ void k_setbias(float* __restrict__ out, const float* __restrict__ bias,
                          int Ncols) {
    int idx = blockIdx.x * blockDim.x + threadIdx.x;
    if (idx < BATCH * Ncols) out[idx] = bias[idx % Ncols];
}

__global__ void k_relu(float* __restrict__ out, int n) {
    int idx = blockIdx.x * blockDim.x + threadIdx.x;
    if (idx < n) out[idx] = fmaxf(out[idx], 0.0f);
}

#define LOAD_SET(A0, A1, A2, A3, BV)                                          \
    do {                                                                      \
        A0 = *(const v16bf*)(ap);                                             \
        A1 = *(const v16bf*)(ap + 512);                                       \
        A2 = *(const v16bf*)(ap + 1024);                                      \
        A3 = *(const v16bf*)(ap + 1536);                                      \
        _Pragma("unroll")                                                     \
        for (int p_ = 0; p_ < 8; ++p_) {                                      \
            int k_ = a_kidx((lane & 16) >> 1, p_);                            \
            BV[2 * p_]     = (__bf16)wp[(size_t)k_ * Ncols];                  \
            BV[2 * p_ + 1] = (__bf16)wp[(size_t)(k_ + 1) * Ncols];            \
        }                                                                     \
        __builtin_prefetch(wp + wStep, 0, 0);                                 \
    } while (0)

#define WMMA4(A0, A1, A2, A3, BV)                                             \
    do {                                                                      \
        acc0 = __builtin_amdgcn_wmma_f32_16x16x32_bf16(false, A0, false, BV,  \
                                                       (short)0, acc0, false, false); \
        acc1 = __builtin_amdgcn_wmma_f32_16x16x32_bf16(false, A1, false, BV,  \
                                                       (short)0, acc1, false, false); \
        acc2 = __builtin_amdgcn_wmma_f32_16x16x32_bf16(false, A2, false, BV,  \
                                                       (short)0, acc2, false, false); \
        acc3 = __builtin_amdgcn_wmma_f32_16x16x32_bf16(false, A3, false, BV,  \
                                                       (short)0, acc3, false, false); \
    } while (0)

#define ADV()                                                                 \
    do { ap += 2048; wp += wStep; } while (0)

__global__ void __launch_bounds__(32)
k_gemm(const __bf16* __restrict__ Ap, const float* __restrict__ Wt,
       float* __restrict__ Out, int K, int Ncols, int nTiles, int kChunk) {
    int bid  = blockIdx.x;
    int nt   = bid % nTiles;
    int kc   = bid / nTiles;
    int lane = threadIdx.x;
    int nBase = nt * 16;
    int kStart = kc * kChunk;              // kChunk multiple of 32
    int kEnd   = min(kStart + kChunk, K);
    int ktBegin = kStart >> 5;
    int ktEnd   = (kEnd + 31) >> 5;
    bool fullN  = (nBase + 16) <= Ncols;   // wave-uniform
    int ktFast  = fullN ? min(ktEnd, K >> 5) : ktBegin;  // guard-free tiles

    v8f acc0 = {}, acc1 = {}, acc2 = {}, acc3 = {};
    const __bf16* ap = Ap + (((size_t)ktBegin * 128 + lane) << 4);
    const float*  wp = Wt + (size_t)kStart * Ncols + nBase + (lane & 15);
    const size_t  wStep = (size_t)32 * Ncols;

    int kt = ktBegin;
    int nFast = ktFast - ktBegin;
    if (nFast > 0) {
        // software-pipelined, guard-free main loop (ping-pong register sets)
        v16bf xa0, xa1, xa2, xa3, xb;      // set X
        v16bf ya0, ya1, ya2, ya3, yb;      // set Y
        LOAD_SET(xa0, xa1, xa2, xa3, xb);  // tile 0
        int i = 1;
        for (; i + 1 < nFast; i += 2) {
            ADV(); LOAD_SET(ya0, ya1, ya2, ya3, yb);   // tile i
            WMMA4(xa0, xa1, xa2, xa3, xb);
            ADV(); LOAD_SET(xa0, xa1, xa2, xa3, xb);   // tile i+1
            WMMA4(ya0, ya1, ya2, ya3, yb);
        }
        if (i < nFast) {
            ADV(); LOAD_SET(ya0, ya1, ya2, ya3, yb);   // last tile
            WMMA4(xa0, xa1, xa2, xa3, xb);
            WMMA4(ya0, ya1, ya2, ya3, yb);
        } else {
            WMMA4(xa0, xa1, xa2, xa3, xb);
        }
        ADV();
        kt = ktFast;
    }

    // guarded remainder: K tail and/or ragged N tile (rare)
    for (; kt < ktEnd; ++kt) {
        v16bf a0 = *(const v16bf*)(ap);
        v16bf a1 = *(const v16bf*)(ap + 512);
        v16bf a2 = *(const v16bf*)(ap + 1024);
        v16bf a3 = *(const v16bf*)(ap + 1536);
        v16bf bv;
        bool nok = (nBase + (lane & 15)) < Ncols;
#pragma unroll
        for (int p = 0; p < 8; ++p) {
            int k0 = a_kidx((lane & 16) >> 1, p);
            float g0 = (nok && (kt << 5) + k0     < K) ? wp[(size_t)k0 * Ncols]       : 0.0f;
            float g1 = (nok && (kt << 5) + k0 + 1 < K) ? wp[(size_t)(k0 + 1) * Ncols] : 0.0f;
            bv[2 * p]     = (__bf16)g0;
            bv[2 * p + 1] = (__bf16)g1;
        }
        WMMA4(a0, a1, a2, a3, bv);
        ADV();
    }

    // C/D layout: lane&15 = N column; VGPR v -> M = v (+8 for upper lanes)
    int ncol = nBase + (lane & 15);
    if (ncol < Ncols) {
        int mB = (lane < 16) ? 0 : 8;
#pragma unroll
        for (int v = 0; v < 8; ++v) {
            atomicAdd(&Out[(size_t)(mB + v)      * Ncols + ncol], acc0[v]);
            atomicAdd(&Out[(size_t)(mB + v + 16) * Ncols + ncol], acc1[v]);
            atomicAdd(&Out[(size_t)(mB + v + 32) * Ncols + ncol], acc2[v]);
            atomicAdd(&Out[(size_t)(mB + v + 48) * Ncols + ncol], acc3[v]);
        }
    }
}

// ---------------------------------------------------------------------------
// Host-side launcher
// ---------------------------------------------------------------------------
static inline size_t alignup(size_t x) { return (x + 255) & ~(size_t)255; }

extern "C" void kernel_launch(void* const* d_in, const int* in_sizes, int n_in,
                              void* d_out, int out_size, void* d_ws, size_t ws_size,
                              hipStream_t stream) {
    const float*     x    = (const float*)d_in[0];
    const long long* edge = (const long long*)d_in[1];
    const float* W1 = (const float*)d_in[2],  *b1 = (const float*)d_in[3];
    const float* W2 = (const float*)d_in[4],  *b2 = (const float*)d_in[5];
    const float* W3 = (const float*)d_in[6],  *b3 = (const float*)d_in[7];
    const float* W4 = (const float*)d_in[8],  *b4 = (const float*)d_in[9];
    const float* FC1_W = (const float*)d_in[10], *FC1_b = (const float*)d_in[11];
    const float* FC2_W = (const float*)d_in[12], *FC2_b = (const float*)d_in[13];
    const float* FC3_W = (const float*)d_in[14], *FC3_b = (const float*)d_in[15];
    const float* FC4_W = (const float*)d_in[16], *FC4_b = (const float*)d_in[17];
    float* out = (float*)d_out;

    // workspace carve-up
    char* p = (char*)d_ws;  size_t off = 0;
    auto take = [&](size_t bytes) { char* r = p + off; off += alignup(bytes); return r; };
    int*    cnt     = (int*)   take(N_NODES * 4);
    int*    cursor  = (int*)   take(N_NODES * 4);
    int*    offs    = (int*)   take((N_NODES + 1) * 4);
    float*  dinv    = (float*) take(N_NODES * 4);
    int*    csr_src = (int*)   take((size_t)E_TOT * 4);
    float*  csr_w   = (float*) take((size_t)E_TOT * 4);
    float*  hA      = (float*) take((size_t)N_NODES * NODE_STRIDE * 4);
    float*  hB      = (float*) take((size_t)N_NODES * NODE_STRIDE * 4);
    __bf16* Apack   = (__bf16*)take((size_t)2500 * 128 * 16 * 2);  // 10.24MB
    float*  fc1     = (float*) take((size_t)BATCH * 1000 * 4);
    float*  fc2     = (float*) take((size_t)BATCH * 500 * 4);
    float*  fc3     = (float*) take((size_t)BATCH * 250 * 4);
    (void)ws_size; (void)n_in; (void)in_sizes; (void)out_size;

    // ---- graph normalization + CSR by destination ----
    k_init <<<(N_NODES + 255) / 256, 256, 0, stream>>>(cnt, cursor);
    k_count<<<(N_EDGES + 255) / 256, 256, 0, stream>>>(edge, cnt);
    k_dinv <<<(N_NODES + 255) / 256, 256, 0, stream>>>(cnt, dinv);
    k_scan <<<1, 1024, 0, stream>>>(cnt, offs);
    k_fill <<<(E_TOT + 255) / 256, 256, 0, stream>>>(edge, offs, cursor, dinv,
                                                     csr_src, csr_w);

    // ---- 4 GCN conv layers (lin -> aggregate(+bias,+relu)) ----
    const int NB = (N_NODES * BATCH + 255) / 256;
    k_lin1 <<<NB, 256, 0, stream>>>(x, W1, hA);
    k_agg  <<<N_NODES, 256, 0, stream>>>(hA, csr_src, csr_w, offs, b1, hB);
    k_lin44<<<NB, 256, 0, stream>>>(hB, W2, hA);
    k_agg  <<<N_NODES, 256, 0, stream>>>(hA, csr_src, csr_w, offs, b2, hB);
    k_lin44<<<NB, 256, 0, stream>>>(hB, W3, hA);
    k_agg  <<<N_NODES, 256, 0, stream>>>(hA, csr_src, csr_w, offs, b3, hB);
    k_lin44<<<NB, 256, 0, stream>>>(hB, W4, hA);
    k_agg  <<<N_NODES, 256, 0, stream>>>(hA, csr_src, csr_w, offs, b4, hB);

    // ---- FC stack via bf16 WMMA ----
    auto gemm = [&](const __bf16* Ap, const float* W, float* O,
                    int K, int N, int kchunk) {
        int nTiles  = (N + 15) / 16;
        int kChunks = (K + kchunk - 1) / kchunk;
        k_gemm<<<dim3(nTiles * kChunks), 32, 0, stream>>>(Ap, W, O, K, N,
                                                          nTiles, kchunk);
    };
    auto apack_blocks = [](int nkt) { return (nkt * 128 + 255) / 256; };

    // FC1: K = 80000, N = 1000  (dominant: 10.2 GFLOP over 320MB weights)
    k_apack_gcn<<<apack_blocks(2500), 256, 0, stream>>>(hB, Apack, 2500);
    k_setbias<<<(BATCH * 1000 + 255) / 256, 256, 0, stream>>>(fc1, FC1_b, 1000);
    gemm(Apack, FC1_W, fc1, 80000, 1000, 1024);        // 63 x 79 waves
    k_relu<<<(BATCH * 1000 + 255) / 256, 256, 0, stream>>>(fc1, BATCH * 1000);

    // FC2: K = 1000, N = 500
    k_apack<<<apack_blocks(32), 256, 0, stream>>>(fc1, Apack, 1000, 32);
    k_setbias<<<(BATCH * 500 + 255) / 256, 256, 0, stream>>>(fc2, FC2_b, 500);
    gemm(Apack, FC2_W, fc2, 1000, 500, 64);
    k_relu<<<(BATCH * 500 + 255) / 256, 256, 0, stream>>>(fc2, BATCH * 500);

    // FC3: K = 500, N = 250
    k_apack<<<apack_blocks(16), 256, 0, stream>>>(fc2, Apack, 500, 16);
    k_setbias<<<(BATCH * 250 + 255) / 256, 256, 0, stream>>>(fc3, FC3_b, 250);
    gemm(Apack, FC3_W, fc3, 500, 250, 64);
    k_relu<<<(BATCH * 250 + 255) / 256, 256, 0, stream>>>(fc3, BATCH * 250);

    // FC4: K = 250, N = 150 (no relu)
    k_apack<<<apack_blocks(8), 256, 0, stream>>>(fc3, Apack, 250, 8);
    k_setbias<<<(BATCH * NUM_CLASSES + 255) / 256, 256, 0, stream>>>(out, FC4_b,
                                                                     NUM_CLASSES);
    gemm(Apack, FC4_W, out, 250, NUM_CLASSES, 32);
}
